// MixtureOfMixers_10179072491667
// MI455X (gfx1250) — matrix operand order repaired and checked by reading, not compiled, not run on hardware
//
#include <hip/hip_runtime.h>

// ---------------------------------------------------------------------------
// MixtureOfMixers on MI455X (gfx1250): bf16 WMMA pipeline.
// B=32, N=196, D=768, E=200, K=20, H=196.  Npad=Hpad=224 (=7*32) so every
// WMMA contraction dim tiles exactly; zero padding is mathematically exact.
// ---------------------------------------------------------------------------

typedef __attribute__((ext_vector_type(8)))  __bf16 v8bf;
typedef __attribute__((ext_vector_type(16))) __bf16 v16bf;
typedef __attribute__((ext_vector_type(8)))  float  v8f;

#define B_   32
#define N_   196
#define D_   768
#define E_   200
#define K_   20
#define H_   196
#define NPAD 224
#define HPAD 224
#define NT2  13      // GEMM2 output column tiles (13*16 = 208 >= 196)

// Branch-free tanh-approx GELU. Native V_TANH_F32 if the builtin exists,
// otherwise tanh(u) = 1 - 2/(1 + 2^(2*log2(e)*u)) via native v_exp_f32/v_rcp_f32.
// Saturation is handled by inf/0 semantics: u>>0 -> exp2=inf -> rcp=0 -> t=1;
// u<<0 -> exp2=0 -> t=-1.  No EXEC-mask games, no libm branches.
__device__ __forceinline__ float gelu_tanh(float x) {
  float u = 0.7978845608028654f * (x + 0.044715f * x * x * x);
#if __has_builtin(__builtin_amdgcn_tanhf)
  float t = __builtin_amdgcn_tanhf(u);
#else
  float e = __builtin_amdgcn_exp2f(2.8853900817779268f * u);   // e^(2u)
  float t = 1.0f - 2.0f * __builtin_amdgcn_rcpf(1.0f + e);
#endif
  return 0.5f * x * (1.0f + t);
}

__device__ __forceinline__ v8f vzero8() {
  v8f v;
#pragma unroll
  for (int r = 0; r < 8; ++r) v[r] = 0.0f;
  return v;
}

// A-fragment (16-bit A 16x32, ISA 7.12.2): lane m = lane&15, half = lane>>4.
// Element j: k = 16*half + j (this file's order, not the ISA table's; B matches). Caller passes base = row + kbase + half*16,
// so we need contiguous chunks [base..base+7] and [base+8..base+15].
__device__ __forceinline__ v16bf load_a_frag(const __bf16* base) {
  v8bf lo = *(const v8bf*)(base);
  v8bf hi = *(const v8bf*)(base + 8);
  return __builtin_shufflevector(lo, hi, 0, 1, 2, 3, 4, 5, 6, 7,
                                         8, 9, 10, 11, 12, 13, 14, 15);
}

// ---------------------------------------------------------------------------
// Kernel 1: router (one block per sample)
// ---------------------------------------------------------------------------
__global__ void router_kernel(const float* __restrict__ x,
                              const float* __restrict__ Wr,
                              float* __restrict__ probs,
                              int* __restrict__ idx,
                              float* __restrict__ wn) {
  __shared__ float xm[D_];
  __shared__ float lg[E_];
  int b = blockIdx.x, tid = threadIdx.x;

  for (int d = tid; d < D_; d += blockDim.x) {
    float s = 0.0f;
    for (int n = 0; n < N_; ++n) s += x[((size_t)b * N_ + n) * D_ + d];
    xm[d] = s * (1.0f / N_);
  }
  __syncthreads();

  for (int e = tid; e < E_; e += blockDim.x) {
    const float* wr = Wr + (size_t)e * D_;
    float s = 0.0f;
    for (int d = 0; d < D_; ++d) s += xm[d] * wr[d];
    lg[e] = s;
  }
  __syncthreads();

  if (tid == 0) {
    float mx = lg[0];
    for (int e = 1; e < E_; ++e) mx = fmaxf(mx, lg[e]);
    float sum = 0.0f;
    for (int e = 0; e < E_; ++e) { float p = expf(lg[e] - mx); lg[e] = p; sum += p; }
    float inv = 1.0f / sum;
    for (int e = 0; e < E_; ++e) { lg[e] *= inv; probs[b * E_ + e] = lg[e]; }
    // top-K, sorted descending, first-index ties (torch.topk semantics)
    float wsum = 0.0f;
    for (int k = 0; k < K_; ++k) {
      int am = 0; float bv = lg[0];
      for (int e = 1; e < E_; ++e) if (lg[e] > bv) { bv = lg[e]; am = e; }
      idx[b * K_ + k] = am; wn[b * K_ + k] = bv; wsum += bv; lg[am] = -1.0f;
    }
    float iw = 1.0f / wsum;
    for (int k = 0; k < K_; ++k) wn[b * K_ + k] *= iw;
  }
}

// ---------------------------------------------------------------------------
// Kernel 2: aux load-balancing loss
// ---------------------------------------------------------------------------
__global__ void aux_kernel(const float* __restrict__ probs,
                           const int* __restrict__ idx,
                           float* __restrict__ out_aux) {
  __shared__ float red[256];
  int tid = threadIdx.x;
  float part = 0.0f;
  for (int e = tid; e < E_; e += 256) {
    float ps = 0.0f; int cnt = 0;
    for (int b = 0; b < B_; ++b) {
      ps  += probs[b * E_ + e];
      cnt += (idx[b * K_] == e) ? 1 : 0;
    }
    part += ps * (float)cnt;
  }
  red[tid] = part; __syncthreads();
  for (int s = 128; s > 0; s >>= 1) {
    if (tid < s) red[tid] += red[tid + s];
    __syncthreads();
  }
  if (tid == 0) *out_aux = (float)E_ * red[0] / ((float)B_ * (float)B_);
}

// ---------------------------------------------------------------------------
// Kernel 3: layernorm over tokens -> bf16 z (B, D, NPAD), one wave per row
// ---------------------------------------------------------------------------
__global__ void ln_z_kernel(const float* __restrict__ x, __bf16* __restrict__ z) {
  int wv = threadIdx.x >> 5, lane = threadIdx.x & 31;
  int row = blockIdx.x * 8 + wv;              // 0 .. B*D-1
  int b = row / D_, d = row - b * D_;
  const float* xp = x + (size_t)b * N_ * D_ + d;

  float s = 0.0f, ss = 0.0f;
  for (int n = lane; n < N_; n += 32) {
    float v = xp[(size_t)n * D_];
    s += v; ss += v * v;
  }
#pragma unroll
  for (int o = 16; o > 0; o >>= 1) {
    s  += __shfl_xor(s, o, 32);
    ss += __shfl_xor(ss, o, 32);
  }
  float mean = s * (1.0f / N_);
  float var  = ss * (1.0f / N_) - mean * mean;   // population var (F.layer_norm)
  float rstd = rsqrtf(var + 1e-5f);

  __bf16* zp = z + (size_t)row * NPAD;
  for (int n = lane; n < NPAD; n += 32) {
    float v = (n < N_) ? (xp[(size_t)n * D_] - mean) * rstd : 0.0f;
    zp[n] = (__bf16)v;
  }
}

// ---------------------------------------------------------------------------
// Kernel 4a: f32 -> bf16 with zero padding: (E,R,C) -> (E,Rp,Cp)
// ---------------------------------------------------------------------------
__global__ void pad_convert_kernel(const float* __restrict__ src,
                                   __bf16* __restrict__ dst,
                                   int E, int R, int C, int Rp, int Cp) {
  size_t total = (size_t)E * Rp * Cp;
  for (size_t i = blockIdx.x * (size_t)blockDim.x + threadIdx.x; i < total;
       i += (size_t)gridDim.x * blockDim.x) {
    int cp = (int)(i % Cp);
    size_t t = i / Cp;
    int rp = (int)(t % Rp);
    int e  = (int)(t / Rp);
    float v = (rp < R && cp < C) ? src[((size_t)e * R + rp) * C + cp] : 0.0f;
    dst[i] = (__bf16)v;
  }
}

// ---------------------------------------------------------------------------
// Kernel 4b: f32 bias zero-pad: (E,C) -> (E,Cp).  Makes the mixer's bias
// loads unconditional (no EXEC-divergent loads inside the GEMM loops).
// ---------------------------------------------------------------------------
__global__ void pad_bias_kernel(const float* __restrict__ src,
                                float* __restrict__ dst,
                                int E, int C, int Cp) {
  int total = E * Cp;
  for (int i = blockIdx.x * blockDim.x + threadIdx.x; i < total;
       i += gridDim.x * blockDim.x) {
    int c = i % Cp, e = i / Cp;
    dst[i] = (c < C) ? src[e * C + c] : 0.0f;
  }
}

// ---------------------------------------------------------------------------
// Kernel 5: expert token-mixer. grid = (D/64, B), 128 threads = 4 waves.
// Wave wv owns D-rows [d0+16*wv, d0+16*wv+16). z-tile staged once in LDS and
// reused for all 20 experts; GEMM2 accumulates the wk-weighted expert outputs
// directly into persistent f32 WMMA accumulators.
// ---------------------------------------------------------------------------
__global__ __launch_bounds__(128) void mixer_kernel(
    const __bf16* __restrict__ z,   const __bf16* __restrict__ W1b,
    const __bf16* __restrict__ W2b, const float* __restrict__ b1p,
    const float* __restrict__ b2p,  const int* __restrict__ idx,
    const float* __restrict__ wn,   __bf16* __restrict__ mixT) {
  __shared__ __align__(16) __bf16 zs[64 * NPAD];         // 28 KB
  __shared__ __align__(16) __bf16 atile[4 * 16 * HPAD];  // 28 KB

  int b = blockIdx.y, d0 = blockIdx.x * 64, tid = threadIdx.x;

  // Stage z tile (64 rows x NPAD bf16), fully coalesced 16B copies.
  {
    const uint4* zsrc = (const uint4*)(z + ((size_t)b * D_ + d0) * NPAD);
    uint4* zdst = (uint4*)zs;
    for (int i = tid; i < 64 * NPAD / 8; i += 128) zdst[i] = zsrc[i];
  }
  __syncthreads();

  int wv = tid >> 5, lane = tid & 31, ml = lane & 15, hf = lane >> 4;
  __bf16* aw = atile + wv * 16 * HPAD;
  const __bf16* zrow = zs + ((size_t)(wv * 16 + ml)) * NPAD + hf * 16;
  const __bf16* arow = aw + (size_t)ml * HPAD + hf * 16;

  v8f   acc2[NT2];
  float b2acc[NT2];
#pragma unroll
  for (int nt = 0; nt < NT2; ++nt) { acc2[nt] = vzero8(); b2acc[nt] = 0.0f; }

  for (int k = 0; k < K_; ++k) {
    int   e  = idx[b * K_ + k];
    float wk = wn[b * K_ + k];

    // Warm the caches for the next expert's weights while we compute this one
    // (global_prefetch_b8; weights are L2-resident, this pulls toward the WGP).
    {
      int en = idx[b * K_ + ((k + 1 < K_) ? (k + 1) : k)];
      const char* pf1 = (const char*)(W1b + (size_t)en * HPAD * NPAD);
      const char* pf2 = (const char*)(W2b + (size_t)en * NPAD * HPAD);
#pragma unroll
      for (int t = 0; t < 3; ++t) {
        size_t off = (size_t)tid * 256 + (size_t)t * 32768;
        __builtin_prefetch(pf1 + off, 0, 1);
        __builtin_prefetch(pf2 + off, 0, 1);
      }
    }

    // ---- GEMM1: a = wk * gelu(z @ W1e^T + b1)  -> bf16 a-tile in LDS ----
    const __bf16* W1e = W1b + (size_t)e * HPAD * NPAD;
    const float*  b1e = b1p + (size_t)e * HPAD;
#pragma unroll 2
    for (int ht = 0; ht < HPAD / 16; ++ht) {
      v8f c1 = vzero8();
      int hcol = ht * 16 + ml;                       // C column == B column
      const __bf16* bp0 = W1e + (size_t)hcol * NPAD + hf * 16;
      float bias = b1e[hcol];                        // zero-padded -> unconditional
#pragma unroll
      for (int kt = 0; kt < NPAD / 32; ++kt) {
        v16bf af = load_a_frag(zrow + kt * 32);
        v16bf bf = *(const v16bf*)(bp0 + kt * 32);
        c1 = __builtin_amdgcn_wmma_f32_16x16x32_bf16(
                 false, af, false, bf, (short)0, c1, false, false);
      }
#pragma unroll
      for (int r = 0; r < 8; ++r) {
        float v = wk * gelu_tanh(c1[r] + bias);
        aw[(size_t)(r + 8 * hf) * HPAD + hcol] = (__bf16)v;   // C row m = r + 8*half
      }
    }
    // Per-wave private a-tile; LDS ops are in-order within a wave -> no barrier.

    // ---- GEMM2: acc2 += (wk*a) @ W2e^T (accumulates across all experts) ----
    const __bf16* W2e = W2b + (size_t)e * NPAD * HPAD;
    const float*  b2e = b2p + (size_t)e * NPAD;
#pragma unroll
    for (int nt = 0; nt < NT2; ++nt) {
      int ncol = nt * 16 + ml;
      const __bf16* bp0 = W2e + (size_t)ncol * HPAD + hf * 16;
      v8f c = acc2[nt];
#pragma unroll
      for (int kt = 0; kt < HPAD / 32; ++kt) {
        v16bf af = load_a_frag(arow + kt * 32);
        v16bf bf = *(const v16bf*)(bp0 + kt * 32);
        c = __builtin_amdgcn_wmma_f32_16x16x32_bf16(
                false, af, false, bf, (short)0, c, false, false);
      }
      acc2[nt] = c;
      b2acc[nt] += wk * b2e[ncol];                   // zero-padded -> unconditional
    }
  }

  // Epilogue: mix[b,d,n] -> mixT[b,n,d] (bf16) for the out-projection GEMM.
#pragma unroll
  for (int nt = 0; nt < NT2; ++nt) {
    int n = nt * 16 + ml;
    if (n < N_) {
#pragma unroll
      for (int r = 0; r < 8; ++r) {
        int d = d0 + wv * 16 + r + 8 * hf;
        mixT[((size_t)b * N_ + n) * D_ + d] = (__bf16)(acc2[nt][r] + b2acc[nt]);
      }
    }
  }
}

// ---------------------------------------------------------------------------
// Kernel 6: out = mixT @ Wo^T + bo.  M = B*N = 6272 (392 tiles), N=K=768.
// One wave per (16-row tile, 16-col tile); grid = (392, 12), 4 waves/block.
// ---------------------------------------------------------------------------
__global__ __launch_bounds__(128) void outproj_kernel(
    const __bf16* __restrict__ mixT, const __bf16* __restrict__ Wob,
    const float* __restrict__ bo, float* __restrict__ out) {
  int tid = threadIdx.x, wv = tid >> 5, lane = tid & 31;
  int ml = lane & 15, hf = lane >> 4;
  int mt = blockIdx.x;               // row tile 0..391
  int nt = blockIdx.y * 4 + wv;      // col tile 0..47
  int dout = nt * 16 + ml;

  const __bf16* arow = mixT + ((size_t)mt * 16 + ml) * D_ + hf * 16;
  const __bf16* brow = Wob + (size_t)dout * D_ + hf * 16;

  v8f acc = vzero8();
#pragma unroll 4
  for (int kt = 0; kt < D_ / 32; ++kt) {
    v16bf af = load_a_frag(arow + kt * 32);
    v16bf bf = *(const v16bf*)(brow + kt * 32);
    acc = __builtin_amdgcn_wmma_f32_16x16x32_bf16(
              false, af, false, bf, (short)0, acc, false, false);
  }
  float bias = bo[dout];
#pragma unroll
  for (int r = 0; r < 8; ++r) {
    int row = mt * 16 + r + 8 * hf;
    out[(size_t)row * D_ + dout] = acc[r] + bias;
  }
}

// ---------------------------------------------------------------------------
// Launch
// ---------------------------------------------------------------------------
extern "C" void kernel_launch(void* const* d_in, const int* in_sizes, int n_in,
                              void* d_out, int out_size, void* d_ws, size_t ws_size,
                              hipStream_t stream) {
  (void)in_sizes; (void)n_in; (void)out_size; (void)ws_size;
  const float* x  = (const float*)d_in[0];
  const float* Wr = (const float*)d_in[1];
  const float* W1 = (const float*)d_in[2];
  const float* b1 = (const float*)d_in[3];
  const float* W2 = (const float*)d_in[4];
  const float* b2 = (const float*)d_in[5];
  const float* Wo = (const float*)d_in[6];
  const float* bo = (const float*)d_in[7];
  float* out = (float*)d_out;

  char* p = (char*)d_ws;
  auto carve = [&](size_t bytes) -> void* {
    void* q = (void*)p;
    p += (bytes + 255) & ~(size_t)255;
    return q;
  };
  float*  probs = (float*)carve((size_t)B_ * E_ * 4);
  int*    idxw  = (int*)carve((size_t)B_ * K_ * 4);
  float*  wnw   = (float*)carve((size_t)B_ * K_ * 4);
  __bf16* z     = (__bf16*)carve((size_t)B_ * D_ * NPAD * 2);          // 11 MB
  __bf16* W1b   = (__bf16*)carve((size_t)E_ * HPAD * NPAD * 2);        // 20 MB
  __bf16* W2b   = (__bf16*)carve((size_t)E_ * NPAD * HPAD * 2);        // 20 MB
  __bf16* Wob   = (__bf16*)carve((size_t)D_ * D_ * 2);                 // 1.2 MB
  __bf16* mixT  = (__bf16*)carve((size_t)B_ * N_ * D_ * 2);            // 9.6 MB
  float*  b1pw  = (float*)carve((size_t)E_ * HPAD * 4);
  float*  b2pw  = (float*)carve((size_t)E_ * NPAD * 4);

  router_kernel<<<B_, 256, 0, stream>>>(x, Wr, probs, idxw, wnw);
  aux_kernel<<<1, 256, 0, stream>>>(probs, idxw, out + (size_t)B_ * N_ * D_);
  ln_z_kernel<<<(B_ * D_) / 8, 256, 0, stream>>>(x, z);
  pad_convert_kernel<<<2048, 256, 0, stream>>>(W1, W1b, E_, H_, N_, HPAD, NPAD);
  pad_convert_kernel<<<2048, 256, 0, stream>>>(W2, W2b, E_, N_, H_, NPAD, HPAD);
  pad_convert_kernel<<<1024, 256, 0, stream>>>(Wo, Wob, 1, D_, D_, D_, D_);
  pad_bias_kernel<<<64, 256, 0, stream>>>(b1, b1pw, E_, H_, HPAD);
  pad_bias_kernel<<<64, 256, 0, stream>>>(b2, b2pw, E_, N_, NPAD);
  mixer_kernel<<<dim3(D_ / 64, B_), 128, 0, stream>>>(z, W1b, W2b, b1pw, b2pw,
                                                      idxw, wnw, mixT);
  outproj_kernel<<<dim3((B_ * N_) / 16, 12), 128, 0, stream>>>(mixT, Wob, bo, out);
}
